// MoELayer_84232898609455
// MI455X (gfx1250) — matrix operand order repaired and checked by reading, not compiled, not run on hardware
//
#include <hip/hip_runtime.h>
#include <hip/hip_bf16.h>
#include <cstdint>

// ---------------- problem constants ----------------
constexpr int Bc = 4, Nc = 2048, Dc = 1024, Hc = 2048, Ec = 8, Oc = 1024, Kc = 2;
constexpr int Tc  = Bc * Nc;      // 8192 tokens
constexpr int TKc = Tc * Kc;      // 16384 routed slots

typedef __attribute__((ext_vector_type(16))) __bf16 v16bf;
typedef __attribute__((ext_vector_type(8)))  float  v8f;

union Frag {
    v16bf v;
    uint4 q[2];
};

__device__ __forceinline__ unsigned short f2bf(float f) {
    unsigned u = __float_as_uint(f);
    unsigned r = u + 0x7FFFu + ((u >> 16) & 1u);   // RNE
    return (unsigned short)(r >> 16);
}

__device__ __forceinline__ float gelu_erf(float x) {
    return 0.5f * x * (1.0f + erff(x * 0.70710678118654752f));
}

// ---- CDNA5 async global->LDS (ASYNCcnt-tracked, direct-to-LDS DMA path) ----
__device__ __forceinline__ void async_ld_b128(unsigned lds_off, const void* gaddr) {
    asm volatile("global_load_async_to_lds_b128 %0, %1, off"
                 :: "v"(lds_off), "v"(gaddr) : "memory");
}
__device__ __forceinline__ void async_wait0() { asm volatile("s_wait_asynccnt 0x0" ::: "memory"); }
__device__ __forceinline__ void async_wait6() { asm volatile("s_wait_asynccnt 0x6" ::: "memory"); }

__device__ __forceinline__ unsigned lds_addr32(const void* p) {
    return (unsigned)(uintptr_t)p;   // LDS aperture: addr[31:0] is the LDS byte offset
}

// ---------------- conversion kernels ----------------
__global__ void cvtx_kernel(const float4* __restrict__ in, ushort4* __restrict__ out, int n4) {
    int i = blockIdx.x * blockDim.x + threadIdx.x;
    if (i >= n4) return;
    float4 v = in[i];
    ushort4 o;
    o.x = f2bf(v.x); o.y = f2bf(v.y); o.z = f2bf(v.z); o.w = f2bf(v.w);
    out[i] = o;
}

// in: [z][R][C] f32  ->  out: [z][C][R] bf16   (R, C multiples of 32)
__global__ void tcvt_kernel(const float* __restrict__ in, unsigned short* __restrict__ out,
                            int R, int C) {
    __shared__ float tile[32][33];
    const float* src = in + (size_t)blockIdx.z * R * C;
    unsigned short* dst = out + (size_t)blockIdx.z * R * C;
    int c0 = blockIdx.x * 32, r0 = blockIdx.y * 32;
    int tx = threadIdx.x, ty = threadIdx.y;                 // block (32,8)
    #pragma unroll
    for (int j = 0; j < 32; j += 8)
        tile[ty + j][tx] = src[(size_t)(r0 + ty + j) * C + (c0 + tx)];
    __syncthreads();
    #pragma unroll
    for (int j = 0; j < 32; j += 8)
        dst[(size_t)(c0 + ty + j) * R + (r0 + tx)] = f2bf(tile[tx][ty + j]);
}

// ---------------- gating ----------------
__global__ void gate_kernel(const float* __restrict__ x, const float* __restrict__ gate_w,
                            const float* __restrict__ gate_b,
                            int* __restrict__ tidx, float* __restrict__ tw,
                            int* __restrict__ counts) {
    __shared__ float gwLDS[Dc * Ec];                        // 32 KB
    int tid = threadIdx.x;
    for (int i = tid; i < Dc * Ec; i += blockDim.x) gwLDS[i] = gate_w[i];
    __syncthreads();

    int t = blockIdx.x * blockDim.x + tid;
    float acc[Ec];
    #pragma unroll
    for (int e = 0; e < Ec; ++e) acc[e] = gate_b[e];

    const float4* xr = (const float4*)(x + (size_t)t * Dc);
    for (int d4 = 0; d4 < Dc / 4; ++d4) {
        float4 xv = xr[d4];
        const float* g = &gwLDS[d4 * 4 * Ec];
        #pragma unroll
        for (int e = 0; e < Ec; ++e) {
            acc[e] += xv.x * g[e] + xv.y * g[Ec + e] + xv.z * g[2 * Ec + e] + xv.w * g[3 * Ec + e];
        }
    }
    // top-2
    int e1 = 0; float l1 = acc[0];
    #pragma unroll
    for (int e = 1; e < Ec; ++e) if (acc[e] > l1) { l1 = acc[e]; e1 = e; }
    int e2 = (e1 == 0) ? 1 : 0; float l2 = acc[e2];
    #pragma unroll
    for (int e = 0; e < Ec; ++e) if (e != e1 && acc[e] > l2) { l2 = acc[e]; e2 = e; }
    float p = __expf(l2 - l1);
    float w1 = 1.0f / (1.0f + p);
    float w2 = p * w1;

    tidx[t * 2 + 0] = e1; tidx[t * 2 + 1] = e2;
    tw[t * 2 + 0] = w1;   tw[t * 2 + 1] = w2;
    atomicAdd(&counts[e1], 1);
    atomicAdd(&counts[e2], 1);
}

__global__ void scan_kernel(const int* __restrict__ counts, int* __restrict__ offsets) {
    int s = 0;
    for (int e = 0; e < Ec; ++e) { offsets[e] = s; s += counts[e]; }
    offsets[Ec] = s;
}

__global__ void scatter_kernel(const int* __restrict__ tidx, const float* __restrict__ tw,
                               const int* __restrict__ offsets, int* __restrict__ cursors,
                               int* __restrict__ tok, float* __restrict__ gws) {
    int t = blockIdx.x * blockDim.x + threadIdx.x;
    if (t >= Tc) return;
    #pragma unroll
    for (int k = 0; k < Kc; ++k) {
        int e = tidx[t * 2 + k];
        int p = atomicAdd(&cursors[e], 1);
        int s = offsets[e] + p;
        tok[s] = t;
        gws[s] = tw[t * 2 + k];
    }
}

// ---------------- WMMA GEMM ----------------
#define WMMA_BF16(ACC, A, B) \
    (ACC) = __builtin_amdgcn_wmma_f32_16x16x32_bf16(false, (A), false, (B), (short)0, (ACC), false, false)

// Block tile 128(M) x 256(N), 8 waves in 2x4 grid, 64x64 per wave (4x4 WMMA accums).
// K-slab 32, double-buffered LDS filled by async global->LDS loads.

// GEMM1: h[slot][H] = gelu( gather(x)[slot] @ w1t[e]^T + b1[e] ), bf16 out
__global__ __launch_bounds__(256)
void gemm1_kernel(const unsigned short* __restrict__ xb,      // [T][D] bf16
                  const unsigned short* __restrict__ w1t,     // [E][H][D] bf16
                  const float* __restrict__ b1,               // [E][H]
                  unsigned short* __restrict__ hb,            // [TK][H] bf16
                  const int* __restrict__ tok,
                  const int* __restrict__ counts, const int* __restrict__ offsets) {
    int e = blockIdx.z;
    int cnt = counts[e];
    int m0 = blockIdx.y * 128;
    if (m0 >= cnt) return;
    int slotBase = offsets[e] + m0;
    int n0 = blockIdx.x * 256;

    __shared__ alignas(16) unsigned short Xs[2][128][40];
    __shared__ alignas(16) unsigned short Ws[2][256][40];

    int tid = threadIdx.x;
    // X loader: row = tid>>1 (0..127), half = tid&1 (16 elems = 32B)
    int xRow = tid >> 1, xHalf = tid & 1;
    int slotL = slotBase + xRow; if (slotL > TKc - 1) slotL = TKc - 1;
    const unsigned short* xrow = xb + (size_t)tok[slotL] * Dc + xHalf * 16;
    unsigned xl[2] = { lds_addr32(&Xs[0][xRow][xHalf * 16]),
                       lds_addr32(&Xs[1][xRow][xHalf * 16]) };
    // W loader: row = tid (0..255), full 32-elem row (64B)
    const unsigned short* wrow = w1t + ((size_t)e * Hc + (n0 + tid)) * Dc;
    unsigned wl[2] = { lds_addr32(&Ws[0][tid][0]), lds_addr32(&Ws[1][tid][0]) };

    auto issue = [&](int bu, int k0) {
        async_ld_b128(xl[bu],      xrow + k0);
        async_ld_b128(xl[bu] + 16, xrow + k0 + 8);
        async_ld_b128(wl[bu],      wrow + k0);
        async_ld_b128(wl[bu] + 16, wrow + k0 + 8);
        async_ld_b128(wl[bu] + 32, wrow + k0 + 16);
        async_ld_b128(wl[bu] + 48, wrow + k0 + 24);
    };

    int lane = tid & 31, wv = tid >> 5;
    int wm = wv >> 2, wn = wv & 3;                 // 2 x 4 wave grid
    int aBase = wm * 64, bBase = wn * 64;
    int l15 = lane & 15, hi = lane >> 4;
    int c0 = hi * 8, kb = hi * 16;

    v8f acc[4][4];
    v8f zero = {0.f, 0.f, 0.f, 0.f, 0.f, 0.f, 0.f, 0.f};
    #pragma unroll
    for (int mi = 0; mi < 4; ++mi)
        #pragma unroll
        for (int ni = 0; ni < 4; ++ni) acc[mi][ni] = zero;

    constexpr int NS = Dc / 32;
    issue(0, 0);
    for (int s = 0; s < NS; ++s) {
        int cur = s & 1;
        if (s + 1 < NS) { issue(cur ^ 1, (s + 1) * 32); async_wait6(); }
        else            { async_wait0(); }
        __syncthreads();                       // current slab visible to all waves

        Frag a[4], b[4];
        #pragma unroll
        for (int mi = 0; mi < 4; ++mi) {
            a[mi].q[0] = *(const uint4*)&Xs[cur][aBase + mi * 16 + l15][c0];
            a[mi].q[1] = *(const uint4*)&Xs[cur][aBase + mi * 16 + l15][c0 + 16];
        }
        #pragma unroll
        for (int ni = 0; ni < 4; ++ni) {
            b[ni].q[0] = *(const uint4*)&Ws[cur][bBase + ni * 16 + l15][c0];
            b[ni].q[1] = *(const uint4*)&Ws[cur][bBase + ni * 16 + l15][c0 + 16];
        }
        #pragma unroll
        for (int mi = 0; mi < 4; ++mi)
            #pragma unroll
            for (int ni = 0; ni < 4; ++ni)
                WMMA_BF16(acc[mi][ni], a[mi].v, b[ni].v);
        __syncthreads();                       // done reading before buffer reuse
    }

    // epilogue: bias + exact GELU -> bf16
    #pragma unroll
    for (int mi = 0; mi < 4; ++mi) {
        #pragma unroll
        for (int ni = 0; ni < 4; ++ni) {
            int col = n0 + bBase + ni * 16 + l15;
            float bias = b1[e * Hc + col];
            #pragma unroll
            for (int r = 0; r < 8; ++r) {
                int mrow = aBase + mi * 16 + r + hi * 8;
                if (m0 + mrow < cnt) {
                    float v = gelu_erf(acc[mi][ni][r] + bias);
                    hb[(size_t)(slotBase + mrow) * Hc + col] = f2bf(v);
                }
            }
        }
    }
}

// GEMM2: out[token] += gate_w * ( h[slot] @ w2t[e]^T + b2[e] )
__global__ __launch_bounds__(256)
void gemm2_kernel(const unsigned short* __restrict__ hb,      // [TK][H] bf16
                  const unsigned short* __restrict__ w2t,     // [E][O][H] bf16
                  const float* __restrict__ b2,               // [E][O]
                  const float* __restrict__ gws,              // [TK]
                  const int* __restrict__ tok,
                  float* __restrict__ out,                    // [T][O]
                  const int* __restrict__ counts, const int* __restrict__ offsets) {
    int e = blockIdx.z;
    int cnt = counts[e];
    int m0 = blockIdx.y * 128;
    if (m0 >= cnt) return;
    int slotBase = offsets[e] + m0;
    int n0 = blockIdx.x * 256;

    __shared__ alignas(16) unsigned short Xs[2][128][40];
    __shared__ alignas(16) unsigned short Ws[2][256][40];

    int tid = threadIdx.x;
    int xRow = tid >> 1, xHalf = tid & 1;
    int slotL = slotBase + xRow; if (slotL > TKc - 1) slotL = TKc - 1;
    const unsigned short* xrow = hb + (size_t)slotL * Hc + xHalf * 16;
    unsigned xl[2] = { lds_addr32(&Xs[0][xRow][xHalf * 16]),
                       lds_addr32(&Xs[1][xRow][xHalf * 16]) };
    const unsigned short* wrow = w2t + ((size_t)e * Oc + (n0 + tid)) * Hc;
    unsigned wl[2] = { lds_addr32(&Ws[0][tid][0]), lds_addr32(&Ws[1][tid][0]) };

    auto issue = [&](int bu, int k0) {
        async_ld_b128(xl[bu],      xrow + k0);
        async_ld_b128(xl[bu] + 16, xrow + k0 + 8);
        async_ld_b128(wl[bu],      wrow + k0);
        async_ld_b128(wl[bu] + 16, wrow + k0 + 8);
        async_ld_b128(wl[bu] + 32, wrow + k0 + 16);
        async_ld_b128(wl[bu] + 48, wrow + k0 + 24);
    };

    int lane = tid & 31, wv = tid >> 5;
    int wm = wv >> 2, wn = wv & 3;
    int aBase = wm * 64, bBase = wn * 64;
    int l15 = lane & 15, hi = lane >> 4;
    int c0 = hi * 8, kb = hi * 16;

    v8f acc[4][4];
    v8f zero = {0.f, 0.f, 0.f, 0.f, 0.f, 0.f, 0.f, 0.f};
    #pragma unroll
    for (int mi = 0; mi < 4; ++mi)
        #pragma unroll
        for (int ni = 0; ni < 4; ++ni) acc[mi][ni] = zero;

    constexpr int NS = Hc / 32;
    issue(0, 0);
    for (int s = 0; s < NS; ++s) {
        int cur = s & 1;
        if (s + 1 < NS) { issue(cur ^ 1, (s + 1) * 32); async_wait6(); }
        else            { async_wait0(); }
        __syncthreads();

        Frag a[4], b[4];
        #pragma unroll
        for (int mi = 0; mi < 4; ++mi) {
            a[mi].q[0] = *(const uint4*)&Xs[cur][aBase + mi * 16 + l15][c0];
            a[mi].q[1] = *(const uint4*)&Xs[cur][aBase + mi * 16 + l15][c0 + 16];
        }
        #pragma unroll
        for (int ni = 0; ni < 4; ++ni) {
            b[ni].q[0] = *(const uint4*)&Ws[cur][bBase + ni * 16 + l15][c0];
            b[ni].q[1] = *(const uint4*)&Ws[cur][bBase + ni * 16 + l15][c0 + 16];
        }
        #pragma unroll
        for (int mi = 0; mi < 4; ++mi)
            #pragma unroll
            for (int ni = 0; ni < 4; ++ni)
                WMMA_BF16(acc[mi][ni], a[mi].v, b[ni].v);
        __syncthreads();
    }

    #pragma unroll
    for (int mi = 0; mi < 4; ++mi) {
        #pragma unroll
        for (int ni = 0; ni < 4; ++ni) {
            int col = n0 + bBase + ni * 16 + l15;
            float bias = b2[e * Oc + col];
            #pragma unroll
            for (int r = 0; r < 8; ++r) {
                int mrow = aBase + mi * 16 + r + hi * 8;
                if (m0 + mrow < cnt) {
                    int slot = slotBase + mrow;
                    float v = (acc[mi][ni][r] + bias) * gws[slot];
                    atomicAdd(&out[(size_t)tok[slot] * Oc + col], v);
                }
            }
        }
    }
}

// ---------------- launch ----------------
extern "C" void kernel_launch(void* const* d_in, const int* in_sizes, int n_in,
                              void* d_out, int out_size, void* d_ws, size_t ws_size,
                              hipStream_t stream) {
    const float* x      = (const float*)d_in[0];
    const float* gate_w = (const float*)d_in[1];
    const float* gate_b = (const float*)d_in[2];
    const float* w1     = (const float*)d_in[3];
    const float* b1     = (const float*)d_in[4];
    const float* w2     = (const float*)d_in[5];
    const float* b2     = (const float*)d_in[6];
    float* out = (float*)d_out;

    // workspace carve-up (256B aligned)
    size_t off = 0;
    auto carve = [&](size_t bytes) { size_t o = off; off = (off + bytes + 255) & ~(size_t)255; return o; };
    char* ws = (char*)d_ws;
    unsigned short* xb   = (unsigned short*)(ws + carve((size_t)Tc * Dc * 2));
    unsigned short* w1t  = (unsigned short*)(ws + carve((size_t)Ec * Hc * Dc * 2));
    unsigned short* w2t  = (unsigned short*)(ws + carve((size_t)Ec * Oc * Hc * 2));
    unsigned short* hb   = (unsigned short*)(ws + carve((size_t)TKc * Hc * 2));
    int*   tok     = (int*)(ws + carve((size_t)TKc * 4));
    float* gws     = (float*)(ws + carve((size_t)TKc * 4));
    int*   tidx    = (int*)(ws + carve((size_t)Tc * 2 * 4));
    float* tw      = (float*)(ws + carve((size_t)Tc * 2 * 4));
    char*  metaB   = ws + carve(256);
    int*   counts  = (int*)metaB;          // 8 ints
    int*   cursors = (int*)(metaB + 32);   // 8 ints
    int*   offsets = (int*)(metaB + 64);   // 9 ints

    // zero accumulation targets + atomic counters (graph-capture-safe memsets)
    hipMemsetAsync(d_out, 0, (size_t)out_size * sizeof(float), stream);
    hipMemsetAsync(metaB, 0, 64, stream);

    // 1) precision conversion / weight transposition (bf16, k-contiguous)
    {
        int n4 = Tc * Dc / 4;
        cvtx_kernel<<<(n4 + 255) / 256, 256, 0, stream>>>((const float4*)x, (ushort4*)xb, n4);
    }
    {
        dim3 blk(32, 8);
        dim3 g1(Hc / 32, Dc / 32, Ec);   // w1 [E][D][H] -> [E][H][D]
        tcvt_kernel<<<g1, blk, 0, stream>>>(w1, w1t, Dc, Hc);
        dim3 g2(Oc / 32, Hc / 32, Ec);   // w2 [E][H][O] -> [E][O][H]
        tcvt_kernel<<<g2, blk, 0, stream>>>(w2, w2t, Hc, Oc);
    }

    // 2) gating: logits, top-2, softmax, counts
    gate_kernel<<<Tc / 256, 256, 0, stream>>>(x, gate_w, gate_b, tidx, tw, counts);

    // 3) offsets + expert-sorted token list
    scan_kernel<<<1, 1, 0, stream>>>(counts, offsets);
    scatter_kernel<<<Tc / 256, 256, 0, stream>>>(tidx, tw, offsets, cursors, tok, gws);

    // 4) expert GEMM1 + GELU  (M = n_e, N = H, K = D)
    {
        dim3 g(Hc / 256, Tc / 128, Ec);  // worst-case M tiles; early-exit on counts
        gemm1_kernel<<<g, 256, 0, stream>>>(xb, w1t, b1, hb, tok, counts, offsets);
    }
    // 5) expert GEMM2 + bias + gate-weighted scatter-add  (M = n_e, N = O, K = H)
    {
        dim3 g(Oc / 256, Tc / 128, Ec);
        gemm2_kernel<<<g, 256, 0, stream>>>(hb, w2t, b2, gws, tok, out, counts, offsets);
    }
}
